// ToyModel_76038101008766
// MI455X (gfx1250) — matrix operand order repaired and checked by reading, not compiled, not run on hardware
//
#include <hip/hip_runtime.h>

// ---------------------------------------------------------------------------
// VQ-VAE forward on MI455X (gfx1250), bf16 WMMA, fully fused per 64-row tile.
// B=131072, F=784(->800 K-pad), H=392(->416), D=28(->32), K=512.
// Round 2: double-buffered WMMA fragment feeds (break s_wait_dscnt-0 chains),
// async global->LDS staging of the W1 slice (ASYNCcnt path).
// ---------------------------------------------------------------------------

typedef __bf16 bf16;
typedef __attribute__((ext_vector_type(16))) __bf16 v16bf;
typedef __attribute__((ext_vector_type(8)))  float  v8f;
typedef __attribute__((ext_vector_type(4)))  unsigned int u32x4;

#define NB    131072
#define NF    784
#define NFP   800   // F padded to 25*32
#define NH    392
#define NHP   416   // H padded to 13*32 (26 N-tiles)
#define NHS   448   // LDS row stride for h (region also holds 512*28 f32)
#define ND    28
#define NDP   32
#define NK    512

__device__ __forceinline__ bf16 f2bf(float f) {
  unsigned u = __builtin_bit_cast(unsigned, f);
  u += 0x7fffu + ((u >> 16) & 1u);               // round-to-nearest-even
  unsigned short s = (unsigned short)(u >> 16);
  bf16 r; __builtin_memcpy(&r, &s, 2);
  return r;
}

// Load a 16-element bf16 fragment as two 16B chunks (p0: 8 elems, p1: 8 elems).
__device__ __forceinline__ v16bf load_frag(const bf16* p0, const bf16* p1) {
  union { v16bf v; u32x4 q[2]; } u;
  u.q[0] = *reinterpret_cast<const u32x4*>(p0);
  u.q[1] = *reinterpret_cast<const u32x4*>(p1);
  return u.v;
}

__device__ __forceinline__ v8f wmma_bf16(v16bf a, v16bf b, v8f c) {
  return __builtin_amdgcn_wmma_f32_16x16x32_bf16(
      /*neg_a=*/false, a, /*neg_b=*/false, b,
      /*c_mod=*/(short)0, c, /*reuse_a=*/false, /*reuse_b=*/false);
}

// Async global -> LDS copy of 16 bytes (GLOBAL_LOAD_ASYNC_TO_LDS_B128, GV).
__device__ __forceinline__ void async_g2l_b128(unsigned lds_addr, const void* gptr) {
  unsigned long long ga = (unsigned long long)(uintptr_t)gptr;
  asm volatile("global_load_async_to_lds_b128 %0, %1, off"
               :: "v"(lds_addr), "v"(ga) : "memory");
}
__device__ __forceinline__ void wait_asynccnt0() {
  asm volatile("s_wait_asynccnt 0x0" ::: "memory");
}

// ------------------------------- prep kernels ------------------------------

__global__ void k_init(float* accum) { accum[0] = 0.f; accum[1] = 0.f; }

__global__ void k_w1t(const float* __restrict__ W1, bf16* __restrict__ W1T) {
  int i = blockIdx.x * 256 + threadIdx.x;            // [NHP][NFP]
  if (i >= NHP * NFP) return;
  int n = i / NFP, k = i % NFP;
  float v = (n < NH && k < NF) ? W1[(size_t)k * NH + n] : 0.f;
  W1T[i] = f2bf(v);
}

__global__ void k_w2t(const float* __restrict__ W2, bf16* __restrict__ W2T) {
  int i = blockIdx.x * 256 + threadIdx.x;            // [NDP][NHP]
  if (i >= NDP * NHP) return;
  int n = i / NHP, k = i % NHP;
  float v = (n < ND && k < NH) ? W2[(size_t)k * ND + n] : 0.f;
  W2T[i] = f2bf(v);
}

__global__ void k_w3t(const float* __restrict__ W3, bf16* __restrict__ W3T) {
  int i = blockIdx.x * 256 + threadIdx.x;            // [NHP][NDP]
  if (i >= NHP * NDP) return;
  int n = i / NDP, k = i % NDP;
  float v = (n < NH && k < ND) ? W3[(size_t)k * NH + n] : 0.f;
  W3T[i] = f2bf(v);
}

__global__ void k_w4t(const float* __restrict__ W4, bf16* __restrict__ W4T) {
  int i = blockIdx.x * 256 + threadIdx.x;            // [NF][NHP]
  if (i >= NF * NHP) return;
  int n = i / NHP, k = i % NHP;
  float v = (k < NH) ? W4[(size_t)k * NF + n] : 0.f;
  W4T[i] = f2bf(v);
}

__global__ void k_prior(const float* __restrict__ prior, bf16* __restrict__ priorB,
                        float* __restrict__ priorF, float* __restrict__ nb) {
  int c = blockIdx.x * 256 + threadIdx.x;
  if (c >= NK) return;
  float s = 0.f;
  for (int j = 0; j < ND; ++j) {
    float v = prior[(size_t)c * ND + j];
    priorF[c * ND + j] = v;
    priorB[c * NDP + j] = f2bf(v);
    s += v * v;
  }
  for (int j = ND; j < NDP; ++j) priorB[c * NDP + j] = f2bf(0.f);
  nb[c] = s;
}

// ------------------------------ encoder kernel -----------------------------
// grid 2048 x 256; each block owns 64 batch rows end-to-end through argmin.

__global__ __launch_bounds__(256) void enc_kernel(
    const float* __restrict__ x,   const bf16* __restrict__ W1T,
    const float* __restrict__ b1,  const bf16* __restrict__ W2T,
    const float* __restrict__ b2,  const float* __restrict__ priorF,
    const float* __restrict__ nb,  int* __restrict__ idxOut,
    float* __restrict__ accum)
{
  __shared__ __align__(16) bf16  sAsl[64 * 32];        // x slice, bf16
  __shared__ __align__(16) bf16  sBsl[NHP * 32];       // W1T slice (async-staged)
  __shared__ __align__(16) bf16  sH[64 * NHS];         // h (reused as f32 codebook)
  __shared__ __align__(16) float sZ[64 * NDP];         // z_enc
  __shared__ float sRedV[256];
  __shared__ int   sRedI[256];
  __shared__ float sS2[64];

  const int tid  = threadIdx.x;
  const int lane = tid & 31, wv = tid >> 5;
  const int mt   = wv >> 1,  nh = wv & 1;              // 4 M-tiles x 2 N-halves
  const int half = lane >> 4, m = lane & 15, nn = lane & 15;
  const int rowBase = blockIdx.x * 64;
  const unsigned ldsBslBase = (unsigned)(uintptr_t)(void*)sBsl;

  // ---- stage 1: h = relu(x @ W1 + b1), K = 800 (25 steps of 32) ----
  v8f acc[13] = {};
  for (int ks = 0; ks < 25; ++ks) {
    const int kk = ks * 32;
    // async-stage B slice (416x32 bf16 = 26624 B) straight into LDS
    #pragma unroll
    for (int s = 0; s < 7; ++s) {
      int i = tid + s * 256;                           // 1664 chunks of 16 B
      if (i < NHP * 4) {
        int rr = i >> 2, part = i & 3;
        async_g2l_b128(ldsBslBase + (unsigned)i * 16,
                       W1T + (size_t)rr * NFP + kk + part * 8);
      }
    }
    // cooperative A-slice load: 64x32, f32 -> bf16, zero-pad k >= 784
    {
      int r = tid >> 2, cb = (tid & 3) * 8;
      const float* src = x + (size_t)(rowBase + r) * NF;
      #pragma unroll
      for (int c = 0; c < 8; ++c) {
        int col = kk + cb + c;
        sAsl[r * 32 + cb + c] = f2bf(col < NF ? src[col] : 0.f);
      }
    }
    wait_asynccnt0();
    __syncthreads();
    if (ks + 1 < 25)
      __builtin_prefetch(W1T + (size_t)(nh * 208) * NFP + kk + 32, 0, 1);
    // A fragment: row = mt*16+m, K runs k0..k0+15, k0 = half*16
    const bf16* ap = sAsl + (mt * 16 + m) * 32 + half * 16;
    v16bf af = load_frag(ap, ap + 8);
    // double-buffered B fragments: load nt+1 while WMMA nt executes
    const bf16* bp0 = sBsl + (nh * 208 + nn) * 32 + half * 16;
    v16bf bcur = load_frag(bp0, bp0 + 8);
    #pragma unroll
    for (int nt = 0; nt < 13; ++nt) {
      v16bf bnxt = bcur;
      if (nt < 12) {
        const bf16* bp = sBsl + (nh * 208 + (nt + 1) * 16 + nn) * 32 + half * 16;
        bnxt = load_frag(bp, bp + 8);
      }
      acc[nt] = wmma_bf16(af, bcur, acc[nt]);
      bcur = bnxt;
    }
    __syncthreads();
  }
  // epilogue: h = relu(acc + b1) -> LDS bf16
  #pragma unroll
  for (int nt = 0; nt < 13; ++nt) {
    int n = nh * 208 + nt * 16 + nn;
    float bias = (n < NH) ? b1[n] : 0.f;
    #pragma unroll
    for (int r = 0; r < 8; ++r) {
      int row = mt * 16 + r + half * 8;
      float v = acc[nt][r] + bias;
      sH[row * NHS + n] = f2bf(v > 0.f ? v : 0.f);
    }
  }
  __syncthreads();

  // ---- stage 2: z = h @ W2 + b2 ; wave (mt, nh) does 16x16, K = 416 ----
  {
    v8f zc = {};
    const int n = nh * 16 + nn;
    const bf16* ap0 = sH + (mt * 16 + m) * NHS + half * 16;
    const bf16* bq0 = W2T + (size_t)n * NHP + half * 16;
    v16bf acur = load_frag(ap0, ap0 + 8);
    v16bf bcur = load_frag(bq0, bq0 + 8);
    #pragma unroll
    for (int ks = 0; ks < 13; ++ks) {
      v16bf anxt = acur, bnxt = bcur;
      if (ks < 12) {
        int kk = (ks + 1) * 32;
        const bf16* ap = ap0 + kk;
        const bf16* bq = bq0 + kk;
        anxt = load_frag(ap, ap + 8);
        bnxt = load_frag(bq, bq + 8);
      }
      zc = wmma_bf16(acur, bcur, zc);
      acur = anxt; bcur = bnxt;
    }
    float bias = (n < ND) ? b2[n] : 0.f;
    #pragma unroll
    for (int r = 0; r < 8; ++r) {
      int row = mt * 16 + r + half * 8;
      sZ[row * NDP + n] = zc[r] + bias;
    }
  }
  __syncthreads();

  // ---- overlay: stage f32 codebook into the (now dead) h region ----
  float* sPF = reinterpret_cast<float*>(sH);           // 512*28 f32 = 57344 B
  {
    const u32x4* src = reinterpret_cast<const u32x4*>(priorF);
    u32x4* dst = reinterpret_cast<u32x4*>(sPF);
    for (int i = tid; i < NK * ND / 4; i += 256) dst[i] = src[i];
  }
  __syncthreads();

  // ---- stage 3: argmin_c (nb[c] - 2 z.p_c); 4 threads/row x 128 codes ----
  {
    int row = tid >> 2, q = tid & 3;
    float z[ND];
    #pragma unroll
    for (int j = 0; j < ND; ++j) z[j] = sZ[row * NDP + j];
    float bestv = 3.0e38f; int besti = 0;
    for (int c = q * 128; c < q * 128 + 128; ++c) {
      const float* p = sPF + c * ND;
      float dot = 0.f;
      #pragma unroll
      for (int j = 0; j < ND; ++j) dot = __builtin_fmaf(p[j], z[j], dot);
      float dv = nb[c] - 2.f * dot;
      if (dv < bestv) { bestv = dv; besti = c; }
    }
    sRedV[tid] = bestv; sRedI[tid] = besti;
    __syncthreads();
    if (q == 0) {
      #pragma unroll
      for (int k = 1; k < 4; ++k)
        if (sRedV[tid + k] < bestv) { bestv = sRedV[tid + k]; besti = sRedI[tid + k]; }
      idxOut[rowBase + row] = besti;
      const float* p = sPF + besti * ND;
      float s2 = 0.f;
      #pragma unroll
      for (int j = 0; j < ND; ++j) { float d = z[j] - p[j]; s2 = __builtin_fmaf(d, d, s2); }
      sS2[row] = s2;
    }
    __syncthreads();
    if (tid == 0) {
      float s = 0.f;
      for (int r = 0; r < 64; ++r) s += sS2[r];
      atomicAdd(accum + 0, s);                         // sum ||z - target||^2
    }
  }
}

// ------------------------------ decoder kernel -----------------------------
// z = prior[idx]; h2 = relu(z@W3+b3); loss_rec += ||x - (h2@W4+b4)||^2 fused.

__global__ __launch_bounds__(256) void dec_kernel(
    const float* __restrict__ x,   const bf16* __restrict__ priorB,
    const int* __restrict__ idxIn, const bf16* __restrict__ W3T,
    const float* __restrict__ b3,  const bf16* __restrict__ W4T,
    const float* __restrict__ b4,  float* __restrict__ accum)
{
  __shared__ __align__(16) bf16 sA0[64 * NDP];         // quantized z, bf16
  __shared__ __align__(16) bf16 sH2[64 * NHS];         // h2, bf16
  __shared__ float sL[256];

  const int tid  = threadIdx.x;
  const int lane = tid & 31, wv = tid >> 5;
  const int mt   = wv >> 1,  nh = wv & 1;
  const int half = lane >> 4, m = lane & 15, nn = lane & 15;
  const int rowBase = blockIdx.x * 64;

  if (tid < 64) {                                      // gather codebook rows
    const u32x4* src = reinterpret_cast<const u32x4*>(
        priorB + (size_t)idxIn[rowBase + tid] * NDP);
    u32x4* dst = reinterpret_cast<u32x4*>(sA0 + tid * NDP);
    dst[0] = src[0]; dst[1] = src[1]; dst[2] = src[2]; dst[3] = src[3];
  }
  __syncthreads();

  // ---- stage 1: h2 = relu(z @ W3 + b3), K = 32, all 13 N-tiles batched ----
  {
    v8f acc1[13] = {};
    const bf16* ap = sA0 + (mt * 16 + m) * NDP + half * 16;
    v16bf af = load_frag(ap, ap + 8);
    const bf16* bq0 = W3T + (size_t)(nh * 208 + nn) * NDP + half * 16;
    v16bf bcur = load_frag(bq0, bq0 + 8);
    #pragma unroll
    for (int nt = 0; nt < 13; ++nt) {
      v16bf bnxt = bcur;
      if (nt < 12) {
        const bf16* bq = W3T + (size_t)(nh * 208 + (nt + 1) * 16 + nn) * NDP + half * 16;
        bnxt = load_frag(bq, bq + 8);
      }
      acc1[nt] = wmma_bf16(af, bcur, acc1[nt]);
      bcur = bnxt;
    }
    #pragma unroll
    for (int nt = 0; nt < 13; ++nt) {
      int n = nh * 208 + nt * 16 + nn;
      float bias = (n < NH) ? b3[n] : 0.f;
      #pragma unroll
      for (int r = 0; r < 8; ++r) {
        int row = mt * 16 + r + half * 8;
        float v = acc1[nt][r] + bias;
        sH2[row * NHS + n] = f2bf(v > 0.f ? v : 0.f);
      }
    }
  }
  __syncthreads();

  // ---- stage 2: x_rec = h2 @ W4 + b4, K = 416, N = 784 (49 tiles) ----
  float lsum = 0.f;
  const int tstart = nh * 25, tcount = nh ? 24 : 25;   // tiles split 25/24
  for (int base = 0; base < tcount; base += 5) {       // 5 N-tiles per chunk
    const int nc = (tcount - base < 5) ? (tcount - base) : 5;
    v8f acc[5] = {};
    for (int ks = 0; ks < 13; ++ks) {
      int kk = ks * 32;
      const bf16* ap = sH2 + (mt * 16 + m) * NHS + kk + half * 16;
      v16bf af = load_frag(ap, ap + 8);
      v16bf bfr[5];                                    // batch loads, one wait
      #pragma unroll
      for (int j = 0; j < 5; ++j) {
        if (j < nc) {
          int n = (tstart + base + j) * 16 + nn;
          const bf16* bp = W4T + (size_t)n * NHP + kk + half * 16;
          bfr[j] = load_frag(bp, bp + 8);
        }
      }
      #pragma unroll
      for (int j = 0; j < 5; ++j)
        if (j < nc) acc[j] = wmma_bf16(af, bfr[j], acc[j]);
    }
    #pragma unroll
    for (int j = 0; j < 5; ++j) {
      if (j < nc) {
        int col = (tstart + base + j) * 16 + nn;
        float bias = b4[col];
        #pragma unroll
        for (int r = 0; r < 8; ++r) {
          int row = rowBase + mt * 16 + r + half * 8;
          float xr = acc[j][r] + bias;
          float d  = x[(size_t)row * NF + col] - xr;
          lsum = __builtin_fmaf(d, d, lsum);
        }
      }
    }
  }
  sL[tid] = lsum;
  __syncthreads();
  if (tid == 0) {
    float s = 0.f;
    for (int i = 0; i < 256; ++i) s += sL[i];
    atomicAdd(accum + 1, s);                           // sum ||x - x_rec||^2
  }
}

__global__ void k_final(const float* __restrict__ accum, float* __restrict__ out) {
  // loss_1 + 0.25*loss_2 = 1.25 * 0.5 * mean(s2) ; loss_rec = mean(rec)
  out[0] = (0.625f * accum[0] + accum[1]) * (1.0f / (float)NB);
}

// --------------------------------- launcher --------------------------------

extern "C" void kernel_launch(void* const* d_in, const int* in_sizes, int n_in,
                              void* d_out, int out_size, void* d_ws, size_t ws_size,
                              hipStream_t stream) {
  (void)in_sizes; (void)n_in; (void)out_size; (void)ws_size;
  const float* x     = (const float*)d_in[0];
  const float* W1    = (const float*)d_in[1];
  const float* b1    = (const float*)d_in[2];
  const float* W2    = (const float*)d_in[3];
  const float* b2    = (const float*)d_in[4];
  const float* W3    = (const float*)d_in[5];
  const float* b3    = (const float*)d_in[6];
  const float* W4    = (const float*)d_in[7];
  const float* b4    = (const float*)d_in[8];
  const float* prior = (const float*)d_in[9];

  char* ws = (char*)d_ws;
  size_t off = 0;
  auto carve = [&](size_t bytes) {
    size_t p = off; off = (off + bytes + 255) & ~(size_t)255; return p;
  };
  float* accum  = (float*)(ws + carve(2 * sizeof(float)));
  int*   idx    = (int*)  (ws + carve((size_t)NB * sizeof(int)));
  bf16*  W1T    = (bf16*) (ws + carve((size_t)NHP * NFP * 2));
  bf16*  W2T    = (bf16*) (ws + carve((size_t)NDP * NHP * 2));
  bf16*  W3T    = (bf16*) (ws + carve((size_t)NHP * NDP * 2));
  bf16*  W4T    = (bf16*) (ws + carve((size_t)NF  * NHP * 2));
  bf16*  priorB = (bf16*) (ws + carve((size_t)NK  * NDP * 2));
  float* priorF = (float*)(ws + carve((size_t)NK  * ND  * 4));
  float* nb     = (float*)(ws + carve((size_t)NK  * 4));

  k_init <<<1, 1, 0, stream>>>(accum);
  k_w1t  <<<(NHP * NFP + 255) / 256, 256, 0, stream>>>(W1, W1T);
  k_w2t  <<<(NDP * NHP + 255) / 256, 256, 0, stream>>>(W2, W2T);
  k_w3t  <<<(NHP * NDP + 255) / 256, 256, 0, stream>>>(W3, W3T);
  k_w4t  <<<(NF  * NHP + 255) / 256, 256, 0, stream>>>(W4, W4T);
  k_prior<<<(NK + 255) / 256, 256, 0, stream>>>(prior, priorB, priorF, nb);

  enc_kernel<<<NB / 64, 256, 0, stream>>>(x, W1T, b1, W2T, b2, priorF, nb, idx, accum);
  dec_kernel<<<NB / 64, 256, 0, stream>>>(x, priorB, idx, W3T, b3, W4T, b4, accum);
  k_final<<<1, 1, 0, stream>>>(accum, (float*)d_out);
}
